// HubnormTripletLoss_62466004353033
// MI455X (gfx1250) — compile-verified, matrix operand order checked by reading, not run on hardware
//
#include <hip/hip_runtime.h>

// ---------------------------------------------------------------------------
// HubnormTripletLoss on MI455X (gfx1250), wave32.
//
// Sinkhorn normalizations are rank-1 scalings: P = diag(r) * K * diag(c),
// K = exp((s-1)/lamb) computed on the fly (never materialized).
//   r_i = 1 / sum_j K_ij c_j   (row pass: v_wmma_f32_16x16x32_f16, B = ones)
//   c_j = 1 / sum_i K_ij r_i   (col pass: coalesced thread-per-column)
// All scaling vectors are kept in log2 space so every streamed element costs
// one v_fma_f32 + one raw v_exp_f32 (no libm range-fixup, no extra muls):
//   K_ij * c_j = exp2(s*SCALE + (log2 c_j - SCALE))
// 11 read-only passes over 256MB => ~2.8GB @ 23.3TB/s ~ 120us, memory-bound.
// ---------------------------------------------------------------------------

typedef __attribute__((ext_vector_type(16))) _Float16 v16h;
typedef __attribute__((ext_vector_type(8)))  float    v8f;
typedef __attribute__((ext_vector_type(4)))  float    v4f;

#define NN 8192
#define MARGIN 0.2f
// exp((s-1)/lamb) == exp2((s-1) * SCALE), SCALE = log2(e)/lamb
#define EXP2_SCALE (83.3333333333333f * 1.44269504088896f)

// Raw v_exp_f32 / v_log_f32 (base-2). Argument range here never needs the
// libm denormal fixup: exponents stay in (-141, 25) and flush-to-zero of
// sub-1e-38 terms is irrelevant to the sums.
__device__ __forceinline__ float exp2_raw(float x) { return __builtin_amdgcn_exp2f(x); }
__device__ __forceinline__ float log2_raw(float x) { return __builtin_amdgcn_logf(x); }

// ---------------------------------------------------------------------------
// c = 1, lcm = log2(c) - SCALE = -SCALE
__global__ void k_init_c(float* __restrict__ c, float* __restrict__ lcm) {
    int j = blockIdx.x * blockDim.x + threadIdx.x;
    if (j < NN) { c[j] = 1.0f; lcm[j] = -EXP2_SCALE; }
}

// ---------------------------------------------------------------------------
// Row pass: m_i = sum_j exp2(s_ij*SCALE + lcm_j);  r_i = 1/m_i,
// lrm_i = log2(r_i) - SCALE = -log2(m_i) - SCALE.
// Block = 128 threads (4 waves) per 16-row strip; wave owns a 2048-col range.
// WMMA with B == ones: every column of D holds the fp32 row sums, and the
// A-operand K-slot permutation is irrelevant. Lane L feeds row (L&15); lanes
// L and L+16 jointly cover 32 consecutive columns (64B contiguous per lane).
__global__ void k_row_pass(const float* __restrict__ sims,
                           const float* __restrict__ lcm,
                           float* __restrict__ r,
                           float* __restrict__ lrm) {
    __shared__ float red[4 * 16];
    const int tid  = threadIdx.x;
    const int wave = tid >> 5;
    const int lane = tid & 31;
    const int half = lane >> 4;
    const int row  = blockIdx.x * 16 + (lane & 15);
    const long long rowOff = (long long)row * NN;

    v16h ones;
    #pragma unroll
    for (int e = 0; e < 16; ++e) ones[e] = (_Float16)1.0f;

    v8f acc = {};
    int colBase = wave * (NN / 4) + half * 16;  // advances by 32 per WMMA
    #pragma unroll 4
    for (int it = 0; it < (NN / 4) / 32; ++it, colBase += 32) {
        const float* ps = sims + rowOff + colBase;
        const float* pl = lcm + colBase;
        v4f s0 = *(const v4f*)(ps +  0);
        v4f s1 = *(const v4f*)(ps +  4);
        v4f s2 = *(const v4f*)(ps +  8);
        v4f s3 = *(const v4f*)(ps + 12);
        v4f l0 = *(const v4f*)(pl +  0);
        v4f l1 = *(const v4f*)(pl +  4);
        v4f l2 = *(const v4f*)(pl +  8);
        v4f l3 = *(const v4f*)(pl + 12);
        v16h a;
        #pragma unroll
        for (int e = 0; e < 4; ++e) {
            a[e +  0] = (_Float16)exp2_raw(fmaf(s0[e], EXP2_SCALE, l0[e]));
            a[e +  4] = (_Float16)exp2_raw(fmaf(s1[e], EXP2_SCALE, l1[e]));
            a[e +  8] = (_Float16)exp2_raw(fmaf(s2[e], EXP2_SCALE, l2[e]));
            a[e + 12] = (_Float16)exp2_raw(fmaf(s3[e], EXP2_SCALE, l3[e]));
        }
        // (neg_a, A, neg_b, B, c_mod, C, reuse_a, reuse_b)
        acc = __builtin_amdgcn_wmma_f32_16x16x32_f16(
            false, a, false, ones, (short)0, acc, false, false);
    }

    // D layout: lanes 0-15, VGPR v -> M=v; lanes 16-31 -> M=8+v. Column 0 of
    // D (lanes 0 and 16) carries all 16 row sums.
    if ((lane & 15) == 0) {
        #pragma unroll
        for (int v = 0; v < 8; ++v) red[wave * 16 + half * 8 + v] = acc[v];
    }
    __syncthreads();
    if (tid < 16) {
        float m = red[tid] + red[16 + tid] + red[32 + tid] + red[48 + tid];
        r[blockIdx.x * 16 + tid]   = 1.0f / m;
        lrm[blockIdx.x * 16 + tid] = -log2_raw(m) - EXP2_SCALE;
    }
}

// ---------------------------------------------------------------------------
// Col pass stage 1: partial[chunk][j] = sum over 256-row chunk of
// exp2(s_ij*SCALE + lrm_i). Fully coalesced; 1 fma + 1 exp + 1 add per elem.
__global__ void k_col_pass(const float* __restrict__ sims,
                           const float* __restrict__ lrm,
                           float* __restrict__ partial) {
    const int j  = blockIdx.x * 256 + threadIdx.x;
    const int i0 = blockIdx.y * (NN / 32);      // 256 rows per chunk
    const float* p = sims + (long long)i0 * NN + j;
    float acc = 0.0f;
    #pragma unroll 4
    for (int t = 0; t < NN / 32; ++t) {
        float li = lrm[i0 + t];                 // uniform -> scalar load
        acc += exp2_raw(fmaf(p[(long long)t * NN], EXP2_SCALE, li));
    }
    partial[(long long)blockIdx.y * NN + j] = acc;
}

// Col pass stage 2: w_j = sum_chunk partial; c_j = 1/w_j, lcm_j = -log2(w)-SCALE.
__global__ void k_col_reduce(const float* __restrict__ partial,
                             float* __restrict__ c,
                             float* __restrict__ lcm) {
    const int j = blockIdx.x * 256 + threadIdx.x;
    float w = 0.0f;
    #pragma unroll
    for (int k = 0; k < 32; ++k) w += partial[(long long)k * NN + j];
    c[j]   = 1.0f / w;
    lcm[j] = -log2_raw(w) - EXP2_SCALE;
}

// ---------------------------------------------------------------------------
// d_k = P_kk = exp2(s_kk*SCALE + lrm_k + (lcm_k + SCALE))
__global__ void k_diag(const float* __restrict__ sims,
                       const float* __restrict__ lrm,
                       const float* __restrict__ lcm,
                       float* __restrict__ d) {
    const int k = blockIdx.x * 256 + threadIdx.x;
    float e = fmaf(sims[(long long)k * NN + k], EXP2_SCALE,
                   lrm[k] + lcm[k] + EXP2_SCALE);
    d[k] = exp2_raw(e);
}

// ---------------------------------------------------------------------------
// Loss: sum_{i!=j} max(P-d_j+M,0) + max(P-d_i+M,0),
// P = exp2(s_ij*SCALE + lrm_i + log2 c_j). Deterministic LDS reduction.
__global__ void k_loss(const float* __restrict__ sims,
                       const float* __restrict__ lrm,
                       const float* __restrict__ lcm,
                       const float* __restrict__ d,
                       double* __restrict__ part) {
    __shared__ double red[256];
    const int j  = blockIdx.x * 256 + threadIdx.x;
    const int i0 = blockIdx.y * 256;
    const float bj = lcm[j] + EXP2_SCALE;       // log2(c_j)
    const float dj = d[j];
    const float* p = sims + (long long)i0 * NN + j;
    float acc = 0.0f;
    for (int t = 0; t < 256; ++t) {
        const int i = i0 + t;
        float P = exp2_raw(fmaf(p[(long long)t * NN], EXP2_SCALE, lrm[i] + bj));
        float h = fmaxf(P - dj + MARGIN, 0.0f) + fmaxf(P - d[i] + MARGIN, 0.0f);
        acc += (i == j) ? 0.0f : h;
    }
    red[threadIdx.x] = (double)acc;
    __syncthreads();
    for (int st = 128; st > 0; st >>= 1) {
        if (threadIdx.x < st) red[threadIdx.x] += red[threadIdx.x + st];
        __syncthreads();
    }
    if (threadIdx.x == 0) part[blockIdx.y * gridDim.x + blockIdx.x] = red[0];
}

__global__ void k_loss_final(const double* __restrict__ part,
                             float* __restrict__ out) {
    __shared__ double red[256];
    double a = 0.0;
    for (int k = threadIdx.x; k < 1024; k += 256) a += part[k];
    red[threadIdx.x] = a;
    __syncthreads();
    for (int st = 128; st > 0; st >>= 1) {
        if (threadIdx.x < st) red[threadIdx.x] += red[threadIdx.x + st];
        __syncthreads();
    }
    if (threadIdx.x == 0) out[0] = (float)red[0];
}

// ---------------------------------------------------------------------------
extern "C" void kernel_launch(void* const* d_in, const int* in_sizes, int n_in,
                              void* d_out, int out_size, void* d_ws, size_t ws_size,
                              hipStream_t stream) {
    (void)in_sizes; (void)n_in; (void)out_size; (void)ws_size;
    const float* sims = (const float*)d_in[0];

    // Workspace layout (all naturally aligned): ~1.2 MB total.
    float*  r        = (float*)d_ws;             // [NN]
    float*  lrm      = r + NN;                   // [NN]  log2(r) - SCALE
    float*  c        = lrm + NN;                 // [NN]
    float*  lcm      = c + NN;                   // [NN]  log2(c) - SCALE
    float*  dg       = lcm + NN;                 // [NN]
    float*  colPart  = dg + NN;                  // [32*NN]
    double* lossPart = (double*)(colPart + 32 * (long long)NN);   // [1024]
    float*  outf     = (float*)d_out;

    k_init_c<<<NN / 256, 256, 0, stream>>>(c, lcm);
    for (int it = 0; it < 5; ++it) {
        k_row_pass<<<NN / 16, 128, 0, stream>>>(sims, lcm, r, lrm);
        k_col_pass<<<dim3(NN / 256, 32), 256, 0, stream>>>(sims, lrm, colPart);
        k_col_reduce<<<NN / 256, 256, 0, stream>>>(colPart, c, lcm);
    }
    k_diag<<<NN / 256, 256, 0, stream>>>(sims, lrm, lcm, dg);
    k_loss<<<dim3(NN / 256, NN / 256), 256, 0, stream>>>(sims, lrm, lcm, dg, lossPart);
    k_loss_final<<<1, 256, 0, stream>>>(lossPart, outf);
}
